// MonotonicNNAlt_66486093742767
// MI455X (gfx1250) — matrix operand order repaired and checked by reading, not compiled, not run on hardware
//
#include <hip/hip_runtime.h>

typedef __attribute__((ext_vector_type(16))) _Float16 v16h;
typedef __attribute__((ext_vector_type(8)))  float    v8f;

#define N_ROWS   500000
#define NFEAT    64
#define KGRP     8
#define HKDIM    64
#define NCOLS    512                 // KGRP*HKDIM
#define NCT      32                  // column tiles of 16
#define NSLABS   (N_ROWS/16)         // 31250 (exact)
#define SPB      64                  // slabs per block = 8 waves x 8 iters
#define TSTRIDE  20                  // padded tile row stride (floats)

// ---------------------------------------------------------------------------
// Kernel 1: w = exp(z), pre-swizzled into B fragments in this kernel's contiguous k order.
// Fragment (ct, kc): B[K][N] with N = ct*16 + (lane&15),
// K = kc*32 + (lane&16 ? 16 : 0) + j  (16 consecutive halves per lane).
// B[K][N] = exp(z[kh=N][i=K]).
// ---------------------------------------------------------------------------
__global__ void prep_wfrag_kernel(const float* __restrict__ z,
                                  _Float16* __restrict__ wfrag)
{
    int tid = blockIdx.x * blockDim.x + threadIdx.x;
    if (tid >= 64 * 32) return;
    int lane = tid & 31;
    int frag = tid >> 5;                 // frag = ct*2 + kc
    int ct   = frag >> 1;
    int kc   = frag & 1;
    int n    = ct * 16 + (lane & 15);
    int kb   = kc * 32 + ((lane & 16) ? 16 : 0);
    _Float16* dst = wfrag + (size_t)tid * 16;
    const float* zr = z + (size_t)n * NFEAT + kb;
#pragma unroll
    for (int j = 0; j < 16; ++j)
        dst[j] = (_Float16)expf(zr[j]);
}

// ---------------------------------------------------------------------------
// Kernel 2: zero the A histogram region of d_out (harness poisons d_out).
// ---------------------------------------------------------------------------
__global__ void zeroA_kernel(float* __restrict__ A)
{
    int i = threadIdx.x;
    if (i < NCOLS) A[i] = 0.0f;
}

__device__ __forceinline__ float swz_swapx16_f(float v)
{
    // ds_swizzle group-of-32: xor_mask=0x10, or=0, and=0x1f -> SWAPX16
    return __int_as_float(__builtin_amdgcn_ds_swizzle(__float_as_int(v), 0x401F));
}
__device__ __forceinline__ int swz_swapx16_i(int v)
{
    return __builtin_amdgcn_ds_swizzle(v, 0x401F);
}

// ---------------------------------------------------------------------------
// Kernel 3: fused GEMM (f16 WMMA, f32 acc) + maxout/argmax + min/argmin + hist
// ---------------------------------------------------------------------------
__launch_bounds__(256)
__global__ void monot_main_kernel(const float* __restrict__ x,
                                  const float* __restrict__ t,
                                  const _Float16* __restrict__ wfrag_g,
                                  float* __restrict__ y,
                                  float* __restrict__ A)
{
    __shared__ v16h     s_wfrag[64 * 32];              // 64 KB B fragments
    __shared__ float    s_t[NCOLS];                    // 2 KB
    __shared__ unsigned s_hist[NCOLS];                 // 2 KB
    __shared__ __align__(16) float s_tile[8][16 * TSTRIDE]; // 10 KB

    const int tid  = threadIdx.x;
    const int lane = tid & 31;
    const int wave = tid >> 5;
    const int r    = lane & 15;     // tile column (GEMM) / row (reduction)
    const int hh   = lane >> 4;     // wave half

    // cooperative LDS fill: B fragments (4096 uint4), t, hist
    {
        const uint4* src = (const uint4*)wfrag_g;
        uint4* dst = (uint4*)s_wfrag;
        for (int i = tid; i < 4096; i += 256) dst[i] = src[i];
        for (int i = tid; i < NCOLS; i += 256) { s_t[i] = t[i]; s_hist[i] = 0u; }
    }
    __syncthreads();

    float* mytile = &s_tile[wave][0];

    for (int it = 0; it < SPB / 8; ++it) {
        const int slab = blockIdx.x * SPB + it * 8 + wave;
        if (slab >= NSLABS) continue;   // wave-uniform guard

        // ---- A fragments: 16x64 f32 -> two 16x32 f16 fragments ----
        // lane<16 holds K {0-15}+kc*32 ; lane>=16 holds K {16-31}+kc*32
        const float4* xr = (const float4*)(x + (size_t)(slab * 16 + r) * NFEAT);
        const int s2 = hh * 4;          // half-select in float4 units (16 floats)
        v16h a0, a1;
        {
            float4 f0 = xr[0 + s2],  f1 = xr[1 + s2];
            float4 f2 = xr[2 + s2],  f3 = xr[3 + s2];
            float4 g0 = xr[8 + s2],  g1 = xr[9 + s2];
            float4 g2 = xr[10 + s2], g3 = xr[11 + s2];
            a0[0]=(_Float16)f0.x;  a0[1]=(_Float16)f0.y;  a0[2]=(_Float16)f0.z;  a0[3]=(_Float16)f0.w;
            a0[4]=(_Float16)f1.x;  a0[5]=(_Float16)f1.y;  a0[6]=(_Float16)f1.z;  a0[7]=(_Float16)f1.w;
            a0[8]=(_Float16)f2.x;  a0[9]=(_Float16)f2.y;  a0[10]=(_Float16)f2.z; a0[11]=(_Float16)f2.w;
            a0[12]=(_Float16)f3.x; a0[13]=(_Float16)f3.y; a0[14]=(_Float16)f3.z; a0[15]=(_Float16)f3.w;
            a1[0]=(_Float16)g0.x;  a1[1]=(_Float16)g0.y;  a1[2]=(_Float16)g0.z;  a1[3]=(_Float16)g0.w;
            a1[4]=(_Float16)g1.x;  a1[5]=(_Float16)g1.y;  a1[6]=(_Float16)g1.z;  a1[7]=(_Float16)g1.w;
            a1[8]=(_Float16)g2.x;  a1[9]=(_Float16)g2.y;  a1[10]=(_Float16)g2.z; a1[11]=(_Float16)g2.w;
            a1[12]=(_Float16)g3.x; a1[13]=(_Float16)g3.y; a1[14]=(_Float16)g3.z; a1[15]=(_Float16)g3.w;
        }

        float ymin = __builtin_inff(); int kmin = 0, hmin = 0;
        float gmax = -__builtin_inff(); int gidx = 0;

        for (int ct = 0; ct < NCT; ++ct) {
            // C init: t depends only on column -> splat per lane
            const float tv = s_t[ct * 16 + r];
            v8f c;
#pragma unroll
            for (int j = 0; j < 8; ++j) c[j] = tv;

            v16h b0 = s_wfrag[(ct * 2 + 0) * 32 + lane];
            v16h b1 = s_wfrag[(ct * 2 + 1) * 32 + lane];

            c = __builtin_amdgcn_wmma_f32_16x16x32_f16(false, a0, false, b0,
                                                       (short)0, c, false, false);
            c = __builtin_amdgcn_wmma_f32_16x16x32_f16(false, a1, false, b1,
                                                       (short)0, c, false, false);

            // spill 16x16 tile: lane half picks rows v / v+8; col = r
#pragma unroll
            for (int v = 0; v < 8; ++v)
                mytile[(v + 8 * hh) * TSTRIDE + r] = c[v];
            __builtin_amdgcn_wave_barrier();   // DS is in-order per wave

            // stream-reduce: row = r, this half scans cols hh*8 .. hh*8+7
            const float4 q0 = *(const float4*)(mytile + r * TSTRIDE + hh * 8);
            const float4 q1 = *(const float4*)(mytile + r * TSTRIDE + hh * 8 + 4);
            __builtin_amdgcn_wave_barrier();   // keep next tile's stores after reads

            const int cb = (ct & 3) * 16 + hh * 8;   // column index within group
            if (q0.x > gmax) { gmax = q0.x; gidx = cb + 0; }
            if (q0.y > gmax) { gmax = q0.y; gidx = cb + 1; }
            if (q0.z > gmax) { gmax = q0.z; gidx = cb + 2; }
            if (q0.w > gmax) { gmax = q0.w; gidx = cb + 3; }
            if (q1.x > gmax) { gmax = q1.x; gidx = cb + 4; }
            if (q1.y > gmax) { gmax = q1.y; gidx = cb + 5; }
            if (q1.z > gmax) { gmax = q1.z; gidx = cb + 6; }
            if (q1.w > gmax) { gmax = q1.w; gidx = cb + 7; }

            if ((ct & 3) == 3) {
                // combine the two halves of this row (SWAPX16), first-max ties
                float om = swz_swapx16_f(gmax);
                int   oi = swz_swapx16_i(gidx);
                bool take = (om > gmax) || (om == gmax && oi < gidx);
                float m  = take ? om : gmax;
                int  idx = take ? oi : gidx;
                if (m < ymin) { ymin = m; kmin = ct >> 2; hmin = idx; }
                gmax = -__builtin_inff(); gidx = 0;
            }
        }

        if (hh == 0) {
            y[slab * 16 + r] = ymin;
            atomicAdd(&s_hist[kmin * HKDIM + hmin], 1u);
        }
    }

    __syncthreads();
    for (int i = tid; i < NCOLS; i += 256) {
        unsigned cnt = s_hist[i];
        if (cnt) atomicAdd(&A[i], (float)cnt);
    }
}

// ---------------------------------------------------------------------------
extern "C" void kernel_launch(void* const* d_in, const int* in_sizes, int n_in,
                              void* d_out, int out_size, void* d_ws, size_t ws_size,
                              hipStream_t stream)
{
    const float* x = (const float*)d_in[0];   // [500000, 64]
    const float* z = (const float*)d_in[1];   // [8, 64, 64]
    const float* t = (const float*)d_in[2];   // [8, 64]
    float* y = (float*)d_out;                 // [500000]
    float* A = y + N_ROWS;                    // [512]
    _Float16* wfrag = (_Float16*)d_ws;        // 64 KB B-fragment scratch

    prep_wfrag_kernel<<<8, 256, 0, stream>>>(z, wfrag);
    zeroA_kernel<<<1, 512, 0, stream>>>(A);

    int blocks = (NSLABS + SPB - 1) / SPB;    // 489
    monot_main_kernel<<<blocks, 256, 0, stream>>>(x, t, wfrag, y, A);
}
